// HCEMoE_24215025614950
// MI455X (gfx1250) — matrix operand order repaired and checked by reading, not compiled, not run on hardware
//
#include <hip/hip_runtime.h>
#include <math.h>

// ---------------------------------------------------------------------------
// MoE (HCEMoE) for MI455X / gfx1250 — bf16 WMMA GEMM pipeline, fp32 I/O.
// All matmuls run through one tiled v_wmma_f32_16x16x32_bf16 kernel with
// fused bias / GELU / residual / gate-scaled accumulation epilogues.
// ---------------------------------------------------------------------------

typedef __attribute__((ext_vector_type(16))) __bf16 v16bf;
typedef __attribute__((ext_vector_type(8)))  float  v8f;

union FragBF { v16bf v; uint4 q[2]; };

#if defined(__has_builtin)
#if __has_builtin(__builtin_amdgcn_cvt_pk_bf16_f32)
#define HCE_HAVE_PK_BF16 1
#endif
#endif

__device__ __forceinline__ unsigned short f2bf_scalar(float f) {
  // round-to-nearest-even fp32 -> bf16
  unsigned int u = __builtin_bit_cast(unsigned int, f);
  u += 0x7FFFu + ((u >> 16) & 1u);
  return (unsigned short)(u >> 16);
}

// pack two fp32 into one dword of two bf16 (lo, hi)
__device__ __forceinline__ unsigned int pack2bf(float lo, float hi) {
#ifdef HCE_HAVE_PK_BF16
  typedef __attribute__((ext_vector_type(2))) __bf16 v2bf;
  v2bf r = __builtin_amdgcn_cvt_pk_bf16_f32(lo, hi);
  return __builtin_bit_cast(unsigned int, r);
#else
  return (unsigned int)f2bf_scalar(lo) | ((unsigned int)f2bf_scalar(hi) << 16);
#endif
}

__device__ __forceinline__ float gelu_exact(float x) {
  return 0.5f * x * (1.0f + erff(x * 0.70710678118654752f));
}

// ---------------------------------------------------------------------------
// C[M,N] = epilogue( A[M,K] @ W[N,K]^T )  (both operands K-contiguous)
//   W element (n,k) at W[n*ldw + k*wks]   (wks=3 handles Conv1d center tap)
//   ACT      : exact-erf GELU
//   RES      : += (*res_coef) * resid[m*N+n]   (res_coef is a DEVICE scalar)
//   RS       : *= rowscale[m*rs_stride]        (per-token gate weight)
//   ACCUM    : C += val  (else C = val)
// Block tile 128x128x64, 256 threads = 8 wave32 waves in a 2x4 grid,
// each wave computes 64x32 via 4x2 fragments; 16 WMMA per staging round.
// ---------------------------------------------------------------------------
template <int ACT, int RES, int RS, int ACCUM>
__global__ __launch_bounds__(256) void gemm_bf16_wmma(
    const float* __restrict__ A, int lda,
    const float* __restrict__ W, int ldw, int wks,
    const float* __restrict__ bias,
    float* __restrict__ C, int N, int K,
    const float* __restrict__ resid, const float* __restrict__ res_coef,
    const float* __restrict__ rowscale, int rs_stride)
{
  __shared__ unsigned short smA[128 * 64];   // [m][k] bf16
  __shared__ unsigned short smB[128 * 64];   // [n][k] bf16

  const int tid  = threadIdx.x;
  const int m0   = blockIdx.y * 128;
  const int n0   = blockIdx.x * 128;
  const int lane = tid & 31;
  const int wave = tid >> 5;
  const int wm   = wave >> 2;     // 0..1  (M direction)
  const int wn   = wave & 3;      // 0..3  (N direction)
  const int lr   = lane & 15;
  const int lh   = lane >> 4;

  v8f acc[4][2];
  const v8f vz = {0.f, 0.f, 0.f, 0.f, 0.f, 0.f, 0.f, 0.f};
#pragma unroll
  for (int f = 0; f < 4; ++f)
#pragma unroll
    for (int g = 0; g < 2; ++g) acc[f][g] = vz;

  for (int k0 = 0; k0 < K; k0 += 64) {
    // ---- stage A tile (fp32 -> bf16), 128 rows x 64 cols, 8 float4/thread
#pragma unroll
    for (int i = 0; i < 8; ++i) {
      int idx = i * 256 + tid;
      int row = idx >> 4;
      int c   = idx & 15;
      const float* gp = A + (size_t)(m0 + row) * lda + (k0 + c * 4);
      if (k0 + 64 < K) __builtin_prefetch(gp + 64, 0, 3);  // global_prefetch_b8
      float4 v = *(const float4*)gp;
      *(uint2*)(&smA[row * 64 + c * 4]) =
          make_uint2(pack2bf(v.x, v.y), pack2bf(v.z, v.w));
    }
    // ---- stage W tile (fp32 -> bf16), with N bounds + optional K-stride
#pragma unroll
    for (int i = 0; i < 8; ++i) {
      int idx = i * 256 + tid;
      int row = idx >> 4;
      int c   = idx & 15;
      int n   = n0 + row;
      unsigned int p0 = 0u, p1 = 0u;
      if (n < N) {
        if (wks == 1) {
          const float* gp = W + (size_t)n * ldw + (k0 + c * 4);
          if (k0 + 64 < K) __builtin_prefetch(gp + 64, 0, 3);
          float4 v = *(const float4*)gp;
          p0 = pack2bf(v.x, v.y);
          p1 = pack2bf(v.z, v.w);
        } else {
          const float* gp = W + (size_t)n * ldw + (size_t)(k0 + c * 4) * wks;
          p0 = pack2bf(gp[0], gp[wks]);
          p1 = pack2bf(gp[2 * wks], gp[3 * wks]);
        }
      }
      *(uint2*)(&smB[row * 64 + c * 4]) = make_uint2(p0, p1);
    }
    __syncthreads();

    // ---- two K=32 halves per staging round
#pragma unroll
    for (int kh = 0; kh < 2; ++kh) {
      const int koff = kh * 32;
      FragBF af[4], bfr[2];
#pragma unroll
      for (int f = 0; f < 4; ++f) {
        // ISA 16-bit A 16x32: lane holds row M=lr; K chunks {lh*8..+7},{+16}
        const unsigned short* p =
            &smA[(wm * 64 + f * 16 + lr) * 64 + koff + lh * 8];
        af[f].q[0] = *(const uint4*)p;
        af[f].q[1] = *(const uint4*)(p + 16);
      }
#pragma unroll
      for (int g = 0; g < 2; ++g) {
        // ISA bf16 B 32x16: lane holds col N=lr; K chunks {lh*8..+7},{+16}
        const unsigned short* p =
            &smB[(wn * 32 + g * 16 + lr) * 64 + koff + lh * 8];
        bfr[g].q[0] = *(const uint4*)p;
        bfr[g].q[1] = *(const uint4*)(p + 16);
      }
#pragma unroll
      for (int f = 0; f < 4; ++f)
#pragma unroll
        for (int g = 0; g < 2; ++g)
          acc[f][g] = __builtin_amdgcn_wmma_f32_16x16x32_bf16(
              false, af[f].v, false, bfr[g].v, (short)0, acc[f][g], false, false);
    }
    __syncthreads();
  }

  float rc = 0.0f;
  if (RES) rc = *res_coef;   // uniform scalar load of device-side res coef

  // ---- epilogue: C/D layout — VGPR r: M = r + 8*lh (per half-wave), N = lr
#pragma unroll
  for (int g = 0; g < 2; ++g) {
    int n = n0 + wn * 32 + g * 16 + lr;
    if (n >= N) continue;
    float bv = bias ? bias[n] : 0.0f;
#pragma unroll
    for (int f = 0; f < 4; ++f) {
#pragma unroll
      for (int r = 0; r < 8; ++r) {
        int m = m0 + wm * 64 + f * 16 + lh * 8 + r;
        float val = acc[f][g][r] + bv;
        if (ACT) val = gelu_exact(val);
        size_t ci = (size_t)m * N + n;
        if (RES) val += rc * resid[ci];
        if (RS)  val *= rowscale[(size_t)m * rs_stride];
        if (ACCUM) C[ci] += val;
        else       C[ci]  = val;
      }
    }
  }
}

// ---------------------------------------------------------------------------
// LayerNorm over D=1024: out = (v - mean)*rsqrt(var+1e-5)*s + b, v = x (+ add)
// one 256-thread block per token
// ---------------------------------------------------------------------------
__global__ __launch_bounds__(256) void ln_kernel(
    const float* __restrict__ x, const float* __restrict__ add,
    const float* __restrict__ s, const float* __restrict__ b,
    float* __restrict__ out)
{
  __shared__ float row[1024];
  __shared__ float r1[256];
  __shared__ float r2[256];
  const int m = blockIdx.x;
  const int t = threadIdx.x;
  float ps = 0.f, pq = 0.f;
  for (int i = t; i < 1024; i += 256) {
    float v = x[(size_t)m * 1024 + i];
    if (add) v += add[(size_t)m * 1024 + i];
    row[i] = v;
    ps += v; pq += v * v;
  }
  r1[t] = ps; r2[t] = pq;
  __syncthreads();
  for (int o = 128; o > 0; o >>= 1) {
    if (t < o) { r1[t] += r1[t + o]; r2[t] += r2[t + o]; }
    __syncthreads();
  }
  float mean = r1[0] * (1.0f / 1024.0f);
  float var  = r2[0] * (1.0f / 1024.0f) - mean * mean;
  float inv  = rsqrtf(var + 1e-5f);
  for (int i = t; i < 1024; i += 256)
    out[(size_t)m * 1024 + i] = (row[i] - mean) * inv * s[i] + b[i];
}

// ---------------------------------------------------------------------------
// router logits: logits[m,e] = hmid[m,:] . w2[e,:] + b2[e]  (H=512, E=8)
// ---------------------------------------------------------------------------
__global__ void logits_kernel(const float* __restrict__ hmid,
                              const float* __restrict__ w2,
                              const float* __restrict__ b2,
                              float* __restrict__ logits, int Bt)
{
  int id = blockIdx.x * blockDim.x + threadIdx.x;
  int m = id >> 3, e = id & 7;
  if (m >= Bt) return;
  float sacc = b2[e];
  const float* hp = hmid + (size_t)m * 512;
  const float* wp = w2 + (size_t)e * 512;
  for (int k = 0; k < 512; k += 4) {
    sacc += hp[k] * wp[k] + hp[k+1] * wp[k+1] + hp[k+2] * wp[k+2] + hp[k+3] * wp[k+3];
  }
  logits[(size_t)m * 8 + e] = sacc;
}

// ---------------------------------------------------------------------------
// softmax over E=8 + top-2 mask, in place on the logits buffer; cnt = sum top2
// ---------------------------------------------------------------------------
__global__ void gates_kernel(float* __restrict__ g, float* __restrict__ cnt, int Bt)
{
  int m = blockIdx.x * blockDim.x + threadIdx.x;
  if (m >= Bt) return;
  float l[8];
  float mx = -3.4e38f;
  for (int e = 0; e < 8; ++e) { l[e] = g[(size_t)m * 8 + e]; mx = fmaxf(mx, l[e]); }
  float s = 0.f;
  for (int e = 0; e < 8; ++e) { l[e] = expf(l[e] - mx); s += l[e]; }
  float is = 1.0f / s;
  for (int e = 0; e < 8; ++e) l[e] *= is;
  int i0 = 0;
  for (int e = 1; e < 8; ++e) if (l[e] > l[i0]) i0 = e;       // ties -> lowest idx
  int i1 = (i0 == 0) ? 1 : 0;
  for (int e = 0; e < 8; ++e) if (e != i0 && l[e] > l[i1]) i1 = e;
  for (int e = 0; e < 8; ++e)
    g[(size_t)m * 8 + e] = (e == i0 || e == i1) ? l[e] : 0.0f;
  cnt[m] = l[i0] + l[i1];
}

// out[m,n] /= (cnt[m] + 1e-8)
__global__ void finalize_kernel(float* __restrict__ out, const float* __restrict__ cnt,
                                int Bt, int O)
{
  int idx = blockIdx.x * blockDim.x + threadIdx.x;
  if (idx >= Bt * O) return;
  out[idx] /= (cnt[idx / O] + 1e-8f);
}

// ---------------------------------------------------------------------------
// host-side GEMM dispatch over the 5 epilogue variants actually used
// ---------------------------------------------------------------------------
static void launch_gemm(hipStream_t stream, int Bt,
                        const float* A, int lda,
                        const float* W, int ldw, int wks,
                        const float* bias, float* C, int N, int K, int act,
                        const float* resid, const float* resc,
                        const float* rs, int rss, int accum)
{
  dim3 grid((N + 127) / 128, Bt / 128);
  dim3 blk(256);
  if (rs) {
    if (accum)
      gemm_bf16_wmma<0, 0, 1, 1><<<grid, blk, 0, stream>>>(
          A, lda, W, ldw, wks, bias, C, N, K, resid, resc, rs, rss);
    else
      gemm_bf16_wmma<0, 0, 1, 0><<<grid, blk, 0, stream>>>(
          A, lda, W, ldw, wks, bias, C, N, K, resid, resc, rs, rss);
  } else if (resid) {
    gemm_bf16_wmma<0, 1, 0, 0><<<grid, blk, 0, stream>>>(
        A, lda, W, ldw, wks, bias, C, N, K, resid, resc, rs, rss);
  } else if (act) {
    gemm_bf16_wmma<1, 0, 0, 0><<<grid, blk, 0, stream>>>(
        A, lda, W, ldw, wks, bias, C, N, K, resid, resc, rs, rss);
  } else {
    gemm_bf16_wmma<0, 0, 0, 0><<<grid, blk, 0, stream>>>(
        A, lda, W, ldw, wks, bias, C, N, K, resid, resc, rs, rss);
  }
}

extern "C" void kernel_launch(void* const* d_in, const int* in_sizes, int n_in,
                              void* d_out, int out_size, void* d_ws, size_t ws_size,
                              hipStream_t stream)
{
  constexpr int D = 1024, O = 1000, E = 8, H = 512, NL = 5, FD = 4096;
  const int Bt = in_sizes[0] / D;   // 8192, multiple of 128

  const float* x = (const float*)d_in[0];
  // router params (inputs 1..10)
  const float* r_in_w  = (const float*)d_in[1];
  const float* r_in_b  = (const float*)d_in[2];
  const float* r_out_w = (const float*)d_in[3];
  const float* r_out_b = (const float*)d_in[4];
  const float* r_ln_s  = (const float*)d_in[5];
  const float* r_ln_b  = (const float*)d_in[6];
  const float* r_w1    = (const float*)d_in[7];
  const float* r_b1    = (const float*)d_in[8];
  const float* r_w2    = (const float*)d_in[9];
  const float* r_b2    = (const float*)d_in[10];

  // workspace arena (fp32)
  float* ws   = (float*)d_ws;
  float* bufA = ws;                                // (B, D)
  float* bufB = bufA + (size_t)Bt * D;             // (B, D)
  float* bufC = bufB + (size_t)Bt * D;             // (B, D)
  float* bufT = bufC + (size_t)Bt * D;             // (B, 4D) — also hmid / conv hidden
  float* gbuf = bufT + (size_t)Bt * FD;            // (B, E) logits -> gates
  float* cbuf = gbuf + (size_t)Bt * E;             // (B,) top-2 weight sum

  float* out = (float*)d_out;

  // ---------------- router ----------------
  // v = x @ in_w[2D:].T + in_b[2D:]
  launch_gemm(stream, Bt, x, D, r_in_w + (size_t)2 * D * D, D, 1, r_in_b + 2 * D,
              bufA, D, D, 0, nullptr, nullptr, nullptr, 0, 0);
  // a = v @ out_w.T + out_b
  launch_gemm(stream, Bt, bufA, D, r_out_w, D, 1, r_out_b,
              bufB, D, D, 0, nullptr, nullptr, nullptr, 0, 0);
  // hr = LN(x + a)
  ln_kernel<<<dim3(Bt), dim3(256), 0, stream>>>(x, bufB, r_ln_s, r_ln_b, bufC);
  // hmid = gelu(hr @ w1.T + b1)
  launch_gemm(stream, Bt, bufC, D, r_w1, D, 1, r_b1,
              bufT, H, D, 1, nullptr, nullptr, nullptr, 0, 0);
  // logits -> softmax -> top-2 gates
  logits_kernel<<<dim3((Bt * E + 255) / 256), dim3(256), 0, stream>>>(bufT, r_w2, r_b2, gbuf, Bt);
  gates_kernel<<<dim3((Bt + 255) / 256), dim3(256), 0, stream>>>(gbuf, cbuf, Bt);

  // ---------------- experts (dense over all tokens; weight traffic dominates)
  int idx = 11;
  bool first = true;
  for (int e = 0; e < E; ++e) {
    const float* rs = gbuf + e;   // per-token gate weight, stride E
    int t = e % 3;
    if (t == 0) {
      // Conv expert: h = gelu(x @ w1[:,:,1].T + b1); out += wi * (h @ w2[:,:,0].T + b2)
      const float* w1 = (const float*)d_in[idx + 0];
      const float* b1 = (const float*)d_in[idx + 1];
      const float* w2 = (const float*)d_in[idx + 2];
      const float* b2 = (const float*)d_in[idx + 3];
      idx += 4;
      launch_gemm(stream, Bt, x, D, w1 + 1, 3 * D, 3, b1,
                  bufT, 512, D, 1, nullptr, nullptr, nullptr, 0, 0);
      launch_gemm(stream, Bt, bufT, 512, w2, 512, 1, b2,
                  out, O, 512, 0, nullptr, nullptr, rs, E, first ? 0 : 1);
    } else if (t == 1) {
      // Attention expert
      const float* in_w  = (const float*)d_in[idx + 0];
      const float* in_b  = (const float*)d_in[idx + 1];
      const float* out_w = (const float*)d_in[idx + 2];
      const float* out_b = (const float*)d_in[idx + 3];
      const float* ln_s  = (const float*)d_in[idx + 4];
      const float* ln_b  = (const float*)d_in[idx + 5];
      const float* fc_w  = (const float*)d_in[idx + 6];
      const float* fc_b  = (const float*)d_in[idx + 7];
      idx += 8;
      launch_gemm(stream, Bt, x, D, in_w + (size_t)2 * D * D, D, 1, in_b + 2 * D,
                  bufA, D, D, 0, nullptr, nullptr, nullptr, 0, 0);
      launch_gemm(stream, Bt, bufA, D, out_w, D, 1, out_b,
                  bufB, D, D, 0, nullptr, nullptr, nullptr, 0, 0);
      ln_kernel<<<dim3(Bt), dim3(256), 0, stream>>>(bufB, x, ln_s, ln_b, bufC);
      launch_gemm(stream, Bt, bufC, D, fc_w, D, 1, fc_b,
                  out, O, D, 0, nullptr, nullptr, rs, E, first ? 0 : 1);
    } else {
      // Deep expert: 5x (Linear(D,4D) -> Linear(4D,D) + res*h), then fc
      const float* l_w1 = (const float*)d_in[idx + 0];
      const float* l_b1 = (const float*)d_in[idx + 1];
      const float* l_w2 = (const float*)d_in[idx + 2];
      const float* l_b2 = (const float*)d_in[idx + 3];
      const float* resp = (const float*)d_in[idx + 4];   // device scalar res_coef
      const float* fc_w = (const float*)d_in[idx + 5];
      const float* fc_b = (const float*)d_in[idx + 6];
      idx += 7;
      const float* hin = x;
      for (int l = 0; l < NL; ++l) {
        // t = hin @ W1_l.T + b1_l
        launch_gemm(stream, Bt, hin, D, l_w1 + (size_t)l * FD * D, D, 1,
                    l_b1 + (size_t)l * FD, bufT, FD, D, 0,
                    nullptr, nullptr, nullptr, 0, 0);
        // hout = t @ W2_l.T + b2_l + res*hin
        float* hout = (l & 1) ? bufB : bufA;
        launch_gemm(stream, Bt, bufT, FD, l_w2 + (size_t)l * D * FD, FD, 1,
                    l_b2 + (size_t)l * D, hout, D, FD, 0,
                    hin, resp, nullptr, 0, 0);
        hin = hout;
      }
      launch_gemm(stream, Bt, hin, D, fc_w, D, 1, fc_b,
                  out, O, D, 0, nullptr, nullptr, rs, E, first ? 0 : 1);
    }
    first = false;
  }

  // out /= (cnt + 1e-8)
  finalize_kernel<<<dim3((Bt * O + 255) / 256), dim3(256), 0, stream>>>(out, cbuf, Bt, O);
}
